// DeCoupleConvv4_40183714021644
// MI455X (gfx1250) — matrix operand order repaired and checked by reading, not compiled, not run on hardware
//
#include <hip/hip_runtime.h>
#include <hip/hip_bf16.h>

typedef __attribute__((ext_vector_type(16))) _Float16 v16h;
typedef __attribute__((ext_vector_type(8)))  _Float16 v8h;
typedef __attribute__((ext_vector_type(8)))  float    v8f;

#define DIM   48
#define HF    127
#define C1    254      // 2*HF
#define C2    508      // 4*HF
#define NBATCH 2
#define HWPX  65536    // 256*256
#define NPIX  131072   // NBATCH*HWPX

// ---------------------------------------------------------------------------
// Kernel 1: in-projection GEMM  y0[b,o,p] = sum_c x[b,c,p] * w_inP[o,c]
// M = pixels (16/tile), N = 254 (pad 256 -> 16 n-tiles), K = 48 (pad 64 -> 2 ksteps)
// ---------------------------------------------------------------------------
__global__ __launch_bounds__(256) void k_gemm_in(const float* __restrict__ x,
                                                 const float* __restrict__ wInP,
                                                 _Float16* __restrict__ y0) {
    __shared__ _Float16 wl[256 * 64];   // [o][c] f16, zero padded
    int tid = threadIdx.x;
    for (int i = tid; i < 256 * 64; i += 256) {
        int o = i >> 6, c = i & 63;
        float v = (o < C1 && c < DIM) ? wInP[o * DIM + c] : 0.0f;
        wl[i] = (_Float16)v;
    }
    __syncthreads();

    int lane = tid & 31, wv = tid >> 5;
    int mtile = blockIdx.x * 8 + wv;
    int p0 = mtile * 16;
    int m = lane & 15, hi = lane >> 4;
    int P = p0 + m;
    int b = P >> 16, pp = P & (HWPX - 1);
    const float* xrow = x + (size_t)b * DIM * HWPX + pp;

    // A fragments: 16-bit A 16x32 layout (ISA 7.12.2)
    v16h a0, a1;
#pragma unroll
    for (int j = 0; j < 8; ++j) {
        int k0 = (j < 4) ? (hi * 8 + 2 * j) : (16 + hi * 8 + 2 * (j - 4));
        a0[2 * j]     = (k0     < DIM) ? (_Float16)xrow[(size_t)k0 * HWPX]       : (_Float16)0.0f;
        a0[2 * j + 1] = (k0 + 1 < DIM) ? (_Float16)xrow[(size_t)(k0 + 1) * HWPX] : (_Float16)0.0f;
        int c1 = 32 + k0;
        a1[2 * j]     = (c1     < DIM) ? (_Float16)xrow[(size_t)c1 * HWPX]       : (_Float16)0.0f;
        a1[2 * j + 1] = (c1 + 1 < DIM) ? (_Float16)xrow[(size_t)(c1 + 1) * HWPX] : (_Float16)0.0f;
    }

    for (int nt = 0; nt < 16; ++nt) {
        int o = nt * 16 + m;            // B-frag / D-frag column
        v16h b0, b1;
#pragma unroll
        for (int r = 0; r < 8; ++r) {
            int k = (r < 4) ? (hi * 8 + 2 * r) : (16 + hi * 8 + 2 * (r - 4));
            b0[2 * r]     = wl[o * 64 + k];
            b0[2 * r + 1] = wl[o * 64 + k + 1];
            b1[2 * r]     = wl[o * 64 + 32 + k];
            b1[2 * r + 1] = wl[o * 64 + 32 + k + 1];
        }
        v8f c = {};
        c = __builtin_amdgcn_wmma_f32_16x16x32_f16(false, a0, false, b0, (short)0, c, false, false);
        c = __builtin_amdgcn_wmma_f32_16x16x32_f16(false, a1, false, b1, (short)0, c, false, false);
        if (o < C1) {
            v8h st;
#pragma unroll
            for (int r = 0; r < 8; ++r) st[r] = (_Float16)c[r];
            int pbase = (p0 & (HWPX - 1)) + hi * 8;   // M rows = 8 consecutive pixels
            *reinterpret_cast<v8h*>(y0 + ((size_t)b * C1 + o) * HWPX + pbase) = st;
        }
    }
}

// ---------------------------------------------------------------------------
// Kernel 2: depthwise 3x3 (ch-mult 2) + relu : y0(254) -> t(508), zero pad
// ---------------------------------------------------------------------------
__global__ __launch_bounds__(256) void k_dw(const _Float16* __restrict__ y0,
                                            const float* __restrict__ wInD,
                                            _Float16* __restrict__ t) {
    long gid = (long)blockIdx.x * 256 + threadIdx.x;   // over NPIX*C2
    int pp = (int)(gid & (HWPX - 1));
    long bc = gid >> 16;                               // b*C2 + oc
    int b = (int)(bc / C2), oc = (int)(bc % C2);
    int ic = oc >> 1;
    int y = pp >> 8, xc = pp & 255;
    const _Float16* src = y0 + ((size_t)b * C1 + ic) * HWPX;
    const float* w = wInD + oc * 9;
    float acc = 0.0f;
#pragma unroll
    for (int dy = -1; dy <= 1; ++dy) {
        int yy = y + dy;
        if ((unsigned)yy >= 256u) continue;
#pragma unroll
        for (int dx = -1; dx <= 1; ++dx) {
            int xx = xc + dx;
            if ((unsigned)xx >= 256u) continue;
            acc += w[(dy + 1) * 3 + (dx + 1)] * (float)src[yy * 256 + xx];
        }
    }
    t[((size_t)b * C2 + oc) * HWPX + pp] = (_Float16)(acc > 0.0f ? acc : 0.0f);
}

// ---------------------------------------------------------------------------
// Kernel 3: channel sum * 1/9      Kernel 4: 3x3 box blur with edge clamp
// ---------------------------------------------------------------------------
__global__ __launch_bounds__(256) void k_csum(const _Float16* __restrict__ t,
                                              float* __restrict__ s) {
    int gid = blockIdx.x * 256 + threadIdx.x;          // over NPIX
    int b = gid >> 16, pp = gid & (HWPX - 1);
    const _Float16* row = t + (size_t)b * C2 * HWPX + pp;
    float acc = 0.0f;
    for (int c = 0; c < C2; ++c) acc += (float)row[(size_t)c * HWPX];
    s[gid] = acc * (1.0f / 9.0f);
}

__global__ __launch_bounds__(256) void k_blur(const float* __restrict__ s,
                                              float* __restrict__ sb) {
    int gid = blockIdx.x * 256 + threadIdx.x;
    int b = gid >> 16, pp = gid & (HWPX - 1);
    int y = pp >> 8, x = pp & 255;
    const float* sp = s + (size_t)b * HWPX;
    float acc = 0.0f;
#pragma unroll
    for (int dy = -1; dy <= 1; ++dy) {
        int yy = y + dy; yy = yy < 0 ? 0 : (yy > 255 ? 255 : yy);
#pragma unroll
        for (int dx = -1; dx <= 1; ++dx) {
            int xx = x + dx; xx = xx < 0 ? 0 : (xx > 255 ? 255 : xx);
            acc += sp[yy * 256 + xx];
        }
    }
    sb[gid] = acc;
}

// ---------------------------------------------------------------------------
// Kernel 5: middle section. Per 32-pixel tile:
//   h = t - blur ; h1 = relu(W_hp1 . h) into LDS ; l1[c] = relu(blur * a_c)
//   shuffle ; l2 = W_lp2 . l1s ; h2 = W_hp2 . h1s ; route into u_l / u_h
// ---------------------------------------------------------------------------
__global__ __launch_bounds__(256) void k_mid(const _Float16* __restrict__ t,
                                             const float* __restrict__ sblur,
                                             const float* __restrict__ wlp1,
                                             const float* __restrict__ whp1,
                                             const float* __restrict__ wlp2,
                                             const float* __restrict__ whp2,
                                             _Float16* __restrict__ ul,
                                             _Float16* __restrict__ uh) {
    __shared__ _Float16 h1[32 * 512];
    __shared__ float sv[32];
    int tid = threadIdx.x;
    int pbase = blockIdx.x * 32;
    if (tid < 32) sv[tid] = sblur[pbase + tid];
    __syncthreads();

    // phase 1: h1 for all 508 channels of 32 pixels
    for (int it = tid; it < 32 * HF; it += 256) {
        int g = it >> 5, p = it & 31;
        int P = pbase + p;
        int b = P >> 16, pp = P & (HWPX - 1);
        float s = sv[p];
        float hv[4];
#pragma unroll
        for (int i = 0; i < 4; ++i)
            hv[i] = (float)t[((size_t)b * C2 + 4 * g + i) * HWPX + pp] - s;
        const float* w = whp1 + g * 16;
#pragma unroll
        for (int o = 0; o < 4; ++o) {
            float acc = w[o * 4 + 0] * hv[0] + w[o * 4 + 1] * hv[1] +
                        w[o * 4 + 2] * hv[2] + w[o * 4 + 3] * hv[3];
            h1[p * 512 + 4 * g + o] = (_Float16)(acc > 0.0f ? acc : 0.0f);
        }
    }
    __syncthreads();

    // phase 2: shuffled group 1x1 + channel routing
    for (int it = tid; it < 32 * HF; it += 256) {
        int g = it >> 5, p = it & 31;
        int P = pbase + p;
        int b = P >> 16, pp = P & (HWPX - 1);
        float s = sv[p];
        float lh[4], ll[4];
#pragma unroll
        for (int k = 0; k < 4; ++k) {
            int cs = 4 * g + k;
            int oc = (cs % HF) * 4 + (cs / HF);   // inverse channel shuffle
            lh[k] = (float)h1[p * 512 + oc];
            const float* wa = wlp1 + (oc >> 2) * 16 + (oc & 3) * 4;
            float a = wa[0] + wa[1] + wa[2] + wa[3];
            float lv = s * a;
            ll[k] = lv > 0.0f ? lv : 0.0f;
        }
        const float* w2l = wlp2 + g * 16;
        const float* w2h = whp2 + g * 16;
#pragma unroll
        for (int o = 0; o < 4; ++o) {
            float l2 = w2l[o * 4 + 0] * ll[0] + w2l[o * 4 + 1] * ll[1] +
                       w2l[o * 4 + 2] * ll[2] + w2l[o * 4 + 3] * ll[3];
            float h2 = w2h[o * 4 + 0] * lh[0] + w2h[o * 4 + 1] * lh[1] +
                       w2h[o * 4 + 2] * lh[2] + w2h[o * 4 + 3] * lh[3];
            int c2 = 4 * g + o;
            size_t off = ((size_t)b * C2 + c2) * HWPX + pp;
            if (c2 < C1) { ul[off] = (_Float16)l2; uh[off] = (_Float16)h2; }
            else         { uh[off] = (_Float16)l2; ul[off] = (_Float16)h2; }
        }
    }
}

// ---------------------------------------------------------------------------
// Kernel 6: grouped 3x3 (groups=127, 4 in -> 1 out) + relu for both paths
//   v channel q*127+g  (q=0: u_l/w_ld, q=1: u_h/w_hd)
// ---------------------------------------------------------------------------
__global__ __launch_bounds__(256) void k_gconv(const _Float16* __restrict__ ul,
                                               const _Float16* __restrict__ uh,
                                               const float* __restrict__ wld,
                                               const float* __restrict__ whd,
                                               _Float16* __restrict__ v) {
    long gid = (long)blockIdx.x * 256 + threadIdx.x;   // over NPIX*C1
    int pp = (int)(gid & (HWPX - 1));
    long rest = gid >> 16;                             // b*C1 + co
    int b = (int)(rest / C1), co = (int)(rest % C1);
    int q = co / HF, g = co % HF;
    const _Float16* u = q ? uh : ul;
    const float* w = (q ? whd : wld) + g * 36;
    int y = pp >> 8, xc = pp & 255;
    float acc = 0.0f;
#pragma unroll
    for (int k = 0; k < 4; ++k) {
        const _Float16* src = u + ((size_t)b * C2 + 4 * g + k) * HWPX;
        const float* wk = w + k * 9;
#pragma unroll
        for (int dy = -1; dy <= 1; ++dy) {
            int yy = y + dy;
            if ((unsigned)yy >= 256u) continue;
#pragma unroll
            for (int dx = -1; dx <= 1; ++dx) {
                int xx = xc + dx;
                if ((unsigned)xx >= 256u) continue;
                acc += wk[(dy + 1) * 3 + (dx + 1)] * (float)src[yy * 256 + xx];
            }
        }
    }
    v[((size_t)b * C1 + co) * HWPX + pp] = (_Float16)(acc > 0.0f ? acc : 0.0f);
}

// ---------------------------------------------------------------------------
// Kernel 7: out-projection GEMM + residual
//   out[b,o,p] = sum_c v[b,c,p]*w_out[o,c] + x[b,o,p]
//   M = pixels, N = 48 (3 n-tiles), K = 254 (pad 256 -> 8 ksteps)
// ---------------------------------------------------------------------------
__global__ __launch_bounds__(256) void k_gemm_out(const _Float16* __restrict__ v,
                                                  const float* __restrict__ wOut,
                                                  const float* __restrict__ x,
                                                  float* __restrict__ out) {
    __shared__ _Float16 wl[48 * 256];   // [o][c] f16, zero padded c>=254
    int tid = threadIdx.x;
    for (int i = tid; i < 48 * 256; i += 256) {
        int o = i >> 8, c = i & 255;
        wl[i] = (_Float16)((c < C1) ? wOut[o * C1 + c] : 0.0f);
    }
    __syncthreads();

    int lane = tid & 31, wv = tid >> 5;
    int mtile = blockIdx.x * 8 + wv;
    int p0 = mtile * 16;
    int m = lane & 15, hi = lane >> 4;
    int P = p0 + m;
    int b = P >> 16, pp = P & (HWPX - 1);
    const _Float16* vrow = v + (size_t)b * C1 * HWPX + pp;

    v16h a[8];
#pragma unroll
    for (int kt = 0; kt < 8; ++kt) {
        int cb = kt * 32;
#pragma unroll
        for (int j = 0; j < 8; ++j) {
            int k0 = (j < 4) ? (hi * 8 + 2 * j) : (16 + hi * 8 + 2 * (j - 4));
            int c0 = cb + k0;
            a[kt][2 * j]     = (c0     < C1) ? vrow[(size_t)c0 * HWPX]       : (_Float16)0.0f;
            a[kt][2 * j + 1] = (c0 + 1 < C1) ? vrow[(size_t)(c0 + 1) * HWPX] : (_Float16)0.0f;
        }
    }
#pragma unroll
    for (int nt = 0; nt < 3; ++nt) {
        int o = nt * 16 + m;
        v8f c = {};
#pragma unroll
        for (int kt = 0; kt < 8; ++kt) {
            int cb = kt * 32;
            v16h bf;
#pragma unroll
            for (int r = 0; r < 8; ++r) {
                int k = (r < 4) ? (hi * 8 + 2 * r) : (16 + hi * 8 + 2 * (r - 4));
                bf[2 * r]     = wl[o * 256 + cb + k];
                bf[2 * r + 1] = wl[o * 256 + cb + k + 1];
            }
            c = __builtin_amdgcn_wmma_f32_16x16x32_f16(false, a[kt], false, bf, (short)0, c, false, false);
        }
        int pbase = (p0 & (HWPX - 1)) + hi * 8;
        const float* xr = x + ((size_t)b * DIM + o) * HWPX + pbase;
        float* orow = out + ((size_t)b * DIM + o) * HWPX + pbase;
#pragma unroll
        for (int r = 0; r < 8; ++r) orow[r] = c[r] + xr[r];
    }
}

// ---------------------------------------------------------------------------
extern "C" void kernel_launch(void* const* d_in, const int* in_sizes, int n_in,
                              void* d_out, int out_size, void* d_ws, size_t ws_size,
                              hipStream_t stream) {
    const float* x     = (const float*)d_in[0];
    const float* wInP  = (const float*)d_in[1];
    const float* wInD  = (const float*)d_in[2];
    const float* wlp1  = (const float*)d_in[3];
    const float* whp1  = (const float*)d_in[4];
    const float* wlp2  = (const float*)d_in[5];
    const float* whp2  = (const float*)d_in[6];
    const float* wld   = (const float*)d_in[7];
    const float* whd   = (const float*)d_in[8];
    const float* wOut  = (const float*)d_in[9];
    float* out = (float*)d_out;

    const size_t SZ_Y0 = (size_t)NPIX * C1 * sizeof(_Float16);  //  66.6 MB
    const size_t SZ_T  = (size_t)NPIX * C2 * sizeof(_Float16);  // 133.2 MB
    const size_t SZ_S  = (size_t)NPIX * sizeof(float);          //   0.5 MB

    char* p = (char*)d_ws;
    _Float16* y0    = (_Float16*)p;  p += SZ_Y0;
    _Float16* t     = (_Float16*)p;  p += SZ_T;
    float*    sbuf  = (float*)p;     p += SZ_S;
    float*    sblur = (float*)p;     p += SZ_S;
    _Float16* ul    = (_Float16*)p;  p += SZ_T;
    _Float16* uh    = (_Float16*)p;  p += SZ_T;
    _Float16* vv    = y0;            // y0 dead after k_dw -> reuse for v

    k_gemm_in <<<NPIX / 128, 256, 0, stream>>>(x, wInP, y0);                 // 1024 blocks
    k_dw      <<<(NPIX * (long)C2) / 256, 256, 0, stream>>>(y0, wInD, t);    // 260096
    k_csum    <<<NPIX / 256, 256, 0, stream>>>(t, sbuf);                     // 512
    k_blur    <<<NPIX / 256, 256, 0, stream>>>(sbuf, sblur);                 // 512
    k_mid     <<<NPIX / 32, 256, 0, stream>>>(t, sblur, wlp1, whp1, wlp2, whp2, ul, uh); // 4096
    k_gconv   <<<(NPIX * (long)C1) / 256, 256, 0, stream>>>(ul, uh, wld, whd, vv);       // 130048
    k_gemm_out<<<NPIX / 128, 256, 0, stream>>>(vv, wOut, x, out);            // 1024
}
